// MS_GWCN_77369540870371
// MI455X (gfx1250) — compile-verified, matrix operand order checked by reading, not run on hardware
//
#include <hip/hip_runtime.h>
#include <hip/hip_bf16.h>

#define N_NODES 50000
#define E_EDGES 400000

typedef __attribute__((ext_vector_type(16))) __bf16 v16bf;
typedef __attribute__((ext_vector_type(8)))  float  v8f;

union FragU { uint4 q[2]; v16bf v; };

// round-to-nearest-even f32->bf16, pure 32-bit ops; pair merged via v_perm_b32
__device__ __forceinline__ unsigned pack_bf16x2(float lo, float hi) {
  unsigned ulo = __float_as_uint(lo), uhi = __float_as_uint(hi);
  unsigned rlo = ulo + 0x7FFFu + ((ulo >> 16) & 1u);
  unsigned rhi = uhi + 0x7FFFu + ((uhi >> 16) & 1u);
  return __builtin_amdgcn_perm(rhi, rlo, 0x07060302u);  // {rhi[31:16], rlo[31:16]}
}

__device__ __forceinline__ uint4 pack_bf16x8(const float* v) {
  uint4 r;
  r.x = pack_bf16x2(v[0], v[1]);
  r.y = pack_bf16x2(v[2], v[3]);
  r.z = pack_bf16x2(v[4], v[5]);
  r.w = pack_bf16x2(v[6], v[7]);
  return r;
}

// ---------------------------------------------------------------------------
// Elementwise / setup kernels
// ---------------------------------------------------------------------------
__global__ void zero_f32v4(float4* __restrict__ p, size_t n4) {
  size_t i = (size_t)blockIdx.x * blockDim.x + threadIdx.x;
  if (i < n4) p[i] = make_float4(0.f, 0.f, 0.f, 0.f);
}

__global__ void relu_f32v4(float4* __restrict__ p, size_t n4) {
  size_t i = (size_t)blockIdx.x * blockDim.x + threadIdx.x;
  if (i < n4) {
    float4 v = p[i];
    p[i] = make_float4(fmaxf(v.x, 0.f), fmaxf(v.y, 0.f), fmaxf(v.z, 0.f), fmaxf(v.w, 0.f));
  }
}

__global__ void deg_kernel(float* __restrict__ deg, const int* __restrict__ dst) {
  int e = blockIdx.x * blockDim.x + threadIdx.x;
  if (e < E_EDGES) atomicAdd(&deg[dst[e]], 1.0f);
}

__global__ void dis_kernel(float* __restrict__ dis, const float* __restrict__ deg) {
  int i = blockIdx.x * blockDim.x + threadIdx.x;
  if (i < N_NODES) {
    float dg = deg[i];
    dis[i] = (dg > 0.0f) ? rsqrtf(fmaxf(dg, 1.0f)) : 0.0f;
  }
}

__global__ void norm_kernel(float* __restrict__ nrm, const int* __restrict__ src,
                            const int* __restrict__ dst, const float* __restrict__ dis) {
  int e = blockIdx.x * blockDim.x + threadIdx.x;
  if (e < E_EDGES) nrm[e] = -dis[src[e]] * dis[dst[e]];
}

// ---------------------------------------------------------------------------
// CSR build: rowptr = exclusive-scan(degree), atomic fill, per-row sort.
// ---------------------------------------------------------------------------
__global__ __launch_bounds__(1024) void scan_block(const float* __restrict__ deg,
                                                   int* __restrict__ rowptr,
                                                   int* __restrict__ partials) {
  __shared__ int sh[1024];
  int tid = threadIdx.x;
  int gid = blockIdx.x * 1024 + tid;
  int v = (gid < N_NODES) ? (int)deg[gid] : 0;
  sh[tid] = v;
  __syncthreads();
  for (int off = 1; off < 1024; off <<= 1) {
    int t = (tid >= off) ? sh[tid - off] : 0;
    __syncthreads();
    sh[tid] += t;
    __syncthreads();
  }
  if (gid < N_NODES) rowptr[gid] = sh[tid] - v;        // block-local exclusive
  if (tid == 1023) partials[blockIdx.x] = sh[1023];
}

__global__ void scan_partials(int* __restrict__ partials, int nb) {
  __shared__ int sh[64];
  int tid = threadIdx.x;
  int v = (tid < nb) ? partials[tid] : 0;
  sh[tid] = v;
  __syncthreads();
  for (int off = 1; off < 64; off <<= 1) {
    int t = (tid >= off) ? sh[tid - off] : 0;
    __syncthreads();
    sh[tid] += t;
    __syncthreads();
  }
  if (tid < nb) partials[tid] = sh[tid] - v;           // exclusive block offsets
}

__global__ __launch_bounds__(1024) void scan_add(int* __restrict__ rowptr,
                                                 const int* __restrict__ partials) {
  int gid = blockIdx.x * 1024 + threadIdx.x;
  if (gid < N_NODES) rowptr[gid] += partials[blockIdx.x];
  if (gid == 0) rowptr[N_NODES] = E_EDGES;
}

__global__ void fill_csr(const int* __restrict__ src, const int* __restrict__ dst,
                         const float* __restrict__ nrm, const int* __restrict__ rowptr,
                         int* __restrict__ fcnt, int* __restrict__ cs, float* __restrict__ cw) {
  int e = blockIdx.x * blockDim.x + threadIdx.x;
  if (e >= E_EDGES) return;
  int d = dst[e];
  int pos = rowptr[d] + atomicAdd(&fcnt[d], 1);
  cs[pos] = src[e];
  cw[pos] = nrm[e];
}

// deterministic row order: insertion-sort each row by src (duplicate srcs carry
// identical weights, so ties cannot change the accumulated value)
__global__ void sort_rows(const int* __restrict__ rowptr, int* __restrict__ cs,
                          float* __restrict__ cw) {
  int n = blockIdx.x * blockDim.x + threadIdx.x;
  if (n >= N_NODES) return;
  int e0 = rowptr[n], e1 = rowptr[n + 1];
  for (int i = e0 + 1; i < e1; ++i) {
    int ks = cs[i];
    float kw = cw[i];
    int j = i - 1;
    while (j >= e0 && cs[j] > ks) { cs[j + 1] = cs[j]; cw[j + 1] = cw[j]; --j; }
    cs[j + 1] = ks;
    cw[j + 1] = kw;
  }
}

// ---------------------------------------------------------------------------
// CSR SpMM, atomic-free: out[n] = coef * sum_e w[e]*in[src[e]]  (- sub[n])
// ---------------------------------------------------------------------------
__global__ void spmm_csr4(float* __restrict__ out, const float* __restrict__ in,
                          const int* __restrict__ rowptr, const int* __restrict__ cs,
                          const float* __restrict__ cw, const float* __restrict__ sub,
                          float coef, int F) {
  int n = blockIdx.x;
  int f4 = blockIdx.y * blockDim.x + threadIdx.x;
  if (f4 >= (F >> 2)) return;
  int e0 = rowptr[n], e1 = rowptr[n + 1];
  float ax = 0.f, ay = 0.f, az = 0.f, aw = 0.f;
  for (int e = e0; e < e1; ++e) {
    int s = cs[e];
    float w = cw[e];
    float4 xv = reinterpret_cast<const float4*>(in + (size_t)s * F)[f4];
    ax = fmaf(w, xv.x, ax);
    ay = fmaf(w, xv.y, ay);
    az = fmaf(w, xv.z, az);
    aw = fmaf(w, xv.w, aw);
  }
  float4 r = make_float4(coef * ax, coef * ay, coef * az, coef * aw);
  if (sub) {
    float4 sv = reinterpret_cast<const float4*>(sub + (size_t)n * F)[f4];
    r.x -= sv.x; r.y -= sv.y; r.z -= sv.z; r.w -= sv.w;
  }
  reinterpret_cast<float4*>(out + (size_t)n * F)[f4] = r;
}

__global__ void spmm_csr(float* __restrict__ out, const float* __restrict__ in,
                         const int* __restrict__ rowptr, const int* __restrict__ cs,
                         const float* __restrict__ cw, const float* __restrict__ sub,
                         float coef, int F) {
  int n = blockIdx.x;
  int f = blockIdx.y * blockDim.x + threadIdx.x;
  if (f >= F) return;
  int e0 = rowptr[n], e1 = rowptr[n + 1];
  float acc = 0.f;
  for (int e = e0; e < e1; ++e)
    acc = fmaf(cw[e], in[(size_t)cs[e] * F + f], acc);
  float r = coef * acc;
  if (sub) r -= sub[(size_t)n * F + f];
  out[(size_t)n * F + f] = r;
}

// ---------------------------------------------------------------------------
// GEMM: H[:, col_off:col_off+fout] (+)= A(N x fin) @ W(fin x fout) [+ bias]
// Block = 8 waves as 4(M) x 2(N); wave tile 32x32 -> 4 WMMA / K-step sharing
// 8 ds_load_b128. Double-buffered LDS; f32 accum via v_wmma_f32_16x16x32_bf16.
// ---------------------------------------------------------------------------
__global__ __launch_bounds__(256) void gemm_bf16_wmma(
    const float* __restrict__ A, const float* __restrict__ W,
    const float* __restrict__ bias, float* __restrict__ H,
    int ldh, int col_off, int fin, int fout, int accumulate)
{
  __shared__ uint4 AsB[2][128 * 4];  // 128 rows x 32 bf16 (4 uint4/row)
  __shared__ uint4 BsB[2][64 * 4];   // transposed: 64 out-features x 32 bf16

  const int tid  = threadIdx.x;
  const int lane = tid & 31;
  const int wv   = tid >> 5;
  const int wr   = wv & 3;           // wave M group (0..3)
  const int wc   = wv >> 2;          // wave N group (0..1)
  const int mbase = blockIdx.y * 128;
  const int nbase = blockIdx.x * 64;
  const int half = lane >> 4;
  const int lrow = lane & 15;
  const bool mfull = (mbase + 128 <= N_NODES);

  const int nc0 = nbase + wc * 32 + lrow;   // output feature, N-subtile 0
  const int nc1 = nc0 + 16;                 // N-subtile 1

  v8f acc00 = {}, acc01 = {}, acc10 = {}, acc11 = {};
  if (accumulate) {
    if (mfull) {
#pragma unroll
      for (int j = 0; j < 8; ++j) {
        int r0 = mbase + wr * 32 + half * 8 + j;
        const float* h0 = H + (size_t)r0 * ldh + col_off;
        const float* h1 = H + (size_t)(r0 + 16) * ldh + col_off;
        acc00[j] = h0[nc0]; acc01[j] = h0[nc1];
        acc10[j] = h1[nc0]; acc11[j] = h1[nc1];
      }
    } else {
#pragma unroll
      for (int j = 0; j < 8; ++j) {
        int r0 = mbase + wr * 32 + half * 8 + j;
        int r1 = r0 + 16;
        if (r0 < N_NODES) {
          const float* h0 = H + (size_t)r0 * ldh + col_off;
          acc00[j] = h0[nc0]; acc01[j] = h0[nc1];
        }
        if (r1 < N_NODES) {
          const float* h1 = H + (size_t)r1 * ldh + col_off;
          acc10[j] = h1[nc0]; acc11[j] = h1[nc1];
        }
      }
    }
  } else {
    float b0 = bias ? bias[nc0] : 0.0f;
    float b1 = bias ? bias[nc1] : 0.0f;
#pragma unroll
    for (int j = 0; j < 8; ++j) { acc00[j] = b0; acc10[j] = b0; acc01[j] = b1; acc11[j] = b1; }
  }

  // --- staging: A thread -> 16 K-floats of one row; B thread -> 8 strided ---
  const int arow = tid >> 1;               // 0..127
  const int akg  = (tid & 1) * 16;         // 0 or 16
  const int grow = mbase + arow;
  const bool rowok = grow < N_NODES;
  const float* ap = A + (size_t)grow * fin;
  const int bnrow = tid >> 2;              // 0..63
  const int bkg   = (tid & 3) * 8;         // 0,8,16,24
  const float* wp = W + nbase + bnrow;
  const bool avec = ((fin & 3) == 0);
  const int ntiles = (fin + 31) >> 5;

  auto loadA = [&](int t, float v[16]) {
    int kb = (t << 5) + akg;
    if (rowok && kb + 16 <= fin) {
      if (avec) {
        const float4* p4 = reinterpret_cast<const float4*>(ap + kb);
#pragma unroll
        for (int q = 0; q < 4; ++q) {
          float4 a = p4[q];
          v[q * 4 + 0] = a.x; v[q * 4 + 1] = a.y; v[q * 4 + 2] = a.z; v[q * 4 + 3] = a.w;
        }
      } else {
#pragma unroll
        for (int j = 0; j < 16; ++j) v[j] = ap[kb + j];
      }
    } else {
#pragma unroll
      for (int j = 0; j < 16; ++j)
        v[j] = (rowok && kb + j < fin) ? ap[kb + j] : 0.0f;
    }
  };
  auto loadB = [&](int t, float v[8]) {
    int kb = (t << 5) + bkg;
    if (kb + 8 <= fin) {
#pragma unroll
      for (int j = 0; j < 8; ++j) v[j] = wp[(size_t)(kb + j) * fout];
    } else {
#pragma unroll
      for (int j = 0; j < 8; ++j)
        v[j] = (kb + j < fin) ? wp[(size_t)(kb + j) * fout] : 0.0f;
    }
  };
  const int asidx = arow * 4 + (akg >> 3);    // two consecutive uint4 slots
  const int bsidx = bnrow * 4 + (bkg >> 3);

  {
    float va[16], vb[8];
    loadA(0, va);
    loadB(0, vb);
    AsB[0][asidx]     = pack_bf16x8(va);
    AsB[0][asidx + 1] = pack_bf16x8(va + 8);
    BsB[0][bsidx]     = pack_bf16x8(vb);
  }

  const int ar = wr * 32 + lrow;           // local A row base
  const int bn = wc * 32 + lrow;           // local B feature base

  for (int t = 0; t < ntiles; ++t) {
    const int buf = t & 1;
    __syncthreads();

    float na[16], nb[8];
    const bool more = (t + 1 < ntiles);
    if (more) {
      loadA(t + 1, na);
      loadB(t + 1, nb);
      if (rowok && t + 2 < ntiles)
        __builtin_prefetch(ap + ((t + 2) << 5) + akg, 0, 3);   // global_prefetch_b8
    }

    FragU fa0, fa1, fb0, fb1;
    fa0.q[0] = AsB[buf][ar * 4 + half];
    fa0.q[1] = AsB[buf][ar * 4 + 2 + half];
    fa1.q[0] = AsB[buf][(ar + 16) * 4 + half];
    fa1.q[1] = AsB[buf][(ar + 16) * 4 + 2 + half];
    fb0.q[0] = BsB[buf][bn * 4 + half];
    fb0.q[1] = BsB[buf][bn * 4 + 2 + half];
    fb1.q[0] = BsB[buf][(bn + 16) * 4 + half];
    fb1.q[1] = BsB[buf][(bn + 16) * 4 + 2 + half];

    acc00 = __builtin_amdgcn_wmma_f32_16x16x32_bf16(false, fa0.v, false, fb0.v,
                                                    (short)0, acc00, false, false);
    acc01 = __builtin_amdgcn_wmma_f32_16x16x32_bf16(false, fa0.v, false, fb1.v,
                                                    (short)0, acc01, false, false);
    acc10 = __builtin_amdgcn_wmma_f32_16x16x32_bf16(false, fa1.v, false, fb0.v,
                                                    (short)0, acc10, false, false);
    acc11 = __builtin_amdgcn_wmma_f32_16x16x32_bf16(false, fa1.v, false, fb1.v,
                                                    (short)0, acc11, false, false);

    if (more) {
      AsB[buf ^ 1][asidx]     = pack_bf16x8(na);
      AsB[buf ^ 1][asidx + 1] = pack_bf16x8(na + 8);
      BsB[buf ^ 1][bsidx]     = pack_bf16x8(nb);
    }
  }

  if (mfull) {
#pragma unroll
    for (int j = 0; j < 8; ++j) {
      int r0 = mbase + wr * 32 + half * 8 + j;
      float* h0 = H + (size_t)r0 * ldh + col_off;
      float* h1 = H + (size_t)(r0 + 16) * ldh + col_off;
      h0[nc0] = acc00[j]; h0[nc1] = acc01[j];
      h1[nc0] = acc10[j]; h1[nc1] = acc11[j];
    }
  } else {
#pragma unroll
    for (int j = 0; j < 8; ++j) {
      int r0 = mbase + wr * 32 + half * 8 + j;
      int r1 = r0 + 16;
      if (r0 < N_NODES) {
        float* h0 = H + (size_t)r0 * ldh + col_off;
        h0[nc0] = acc00[j]; h0[nc1] = acc01[j];
      }
      if (r1 < N_NODES) {
        float* h1 = H + (size_t)r1 * ldh + col_off;
        h1[nc0] = acc10[j]; h1[nc1] = acc11[j];
      }
    }
  }
}

// Final FC: out(N x 9) = h(N x 768) @ fc_w(768 x 9) + fc_b
__global__ void fc_kernel(const float* __restrict__ h, const float* __restrict__ w,
                          const float* __restrict__ b, float* __restrict__ out) {
  int node = blockIdx.x * blockDim.x + threadIdx.x;
  if (node >= N_NODES) return;
  float acc[9];
#pragma unroll
  for (int c = 0; c < 9; ++c) acc[c] = b[c];
  const float4* hr = reinterpret_cast<const float4*>(h + (size_t)node * 768);
  for (int k4 = 0; k4 < 192; ++k4) {
    float4 xv = hr[k4];
    const float* wr_ = w + k4 * 4 * 9;
#pragma unroll
    for (int c = 0; c < 9; ++c)
      acc[c] += xv.x * wr_[c] + xv.y * wr_[9 + c] + xv.z * wr_[18 + c] + xv.w * wr_[27 + c];
  }
#pragma unroll
  for (int c = 0; c < 9; ++c) out[(size_t)node * 9 + c] = acc[c];
}

// ---------------------------------------------------------------------------
extern "C" void kernel_launch(void* const* d_in, const int* in_sizes, int n_in,
                              void* d_out, int out_size, void* d_ws, size_t ws_size,
                              hipStream_t stream) {
  (void)in_sizes; (void)n_in; (void)out_size; (void)ws_size;
  const float* x  = (const float*)d_in[0];
  const int*   ei = (const int*)d_in[1];
  const int* src = ei;
  const int* dst = ei + E_EDGES;
  const int wIdx[4] = {2, 8, 14, 20};
  const int bIdx[4] = {5, 11, 17, 23};
  const float* fc_w = (const float*)d_in[26];
  const float* fc_b = (const float*)d_in[27];
  float* out = (float*)d_out;

  // workspace layout (bytes), 256B-aligned regions
  char* wsb = (char*)d_ws;
  size_t o = 0;
  auto wsalloc = [&](size_t bytes) { void* p = wsb + o; o += (bytes + 255) & ~(size_t)255; return p; };
  float* deg     = (float*)wsalloc(N_NODES * 4);
  float* dis     = (float*)wsalloc(N_NODES * 4);
  float* nrm     = (float*)wsalloc(E_EDGES * 4);
  int*   rowptr  = (int*)wsalloc((N_NODES + 1) * 4);
  int*   partials= (int*)wsalloc(64 * 4);
  int*   fcnt    = (int*)wsalloc(N_NODES * 4);
  int*   csr_s   = (int*)wsalloc(E_EDGES * 4);
  float* csr_w   = (float*)wsalloc(E_EDGES * 4);
  float* TB[3];
  TB[0] = (float*)wsalloc((size_t)N_NODES * 768 * 4);
  TB[1] = (float*)wsalloc((size_t)N_NODES * 768 * 4);
  TB[2] = (float*)wsalloc((size_t)N_NODES * 768 * 4);
  float* hA = (float*)wsalloc((size_t)N_NODES * 768 * 4);
  float* hB = (float*)wsalloc((size_t)N_NODES * 768 * 4);

  // normalization + CSR build (deterministic)
  zero_f32v4<<<(N_NODES / 4 + 255) / 256, 256, 0, stream>>>((float4*)deg, N_NODES / 4);
  deg_kernel<<<(E_EDGES + 255) / 256, 256, 0, stream>>>(deg, dst);
  dis_kernel<<<(N_NODES + 255) / 256, 256, 0, stream>>>(dis, deg);
  norm_kernel<<<(E_EDGES + 255) / 256, 256, 0, stream>>>(nrm, src, dst, dis);

  const int nblk = (N_NODES + 1023) / 1024;    // 49
  scan_block<<<nblk, 1024, 0, stream>>>(deg, rowptr, partials);
  scan_partials<<<1, 64, 0, stream>>>(partials, nblk);
  scan_add<<<nblk, 1024, 0, stream>>>(rowptr, partials);
  zero_f32v4<<<(N_NODES / 4 + 255) / 256, 256, 0, stream>>>((float4*)fcnt, N_NODES / 4);
  fill_csr<<<(E_EDGES + 255) / 256, 256, 0, stream>>>(src, dst, nrm, rowptr, fcnt, csr_s, csr_w);
  sort_rows<<<(N_NODES + 255) / 256, 256, 0, stream>>>(rowptr, csr_s, csr_w);

  auto launch_spmm = [&](float* outp, const float* inp, const float* subp,
                         float coef, int fin) {
    if ((fin & 3) == 0) {
      int F4 = fin >> 2;
      int bs = (F4 <= 64) ? 64 : (F4 <= 128 ? 128 : (F4 <= 192 ? 192 : 256));
      dim3 g(N_NODES, (F4 + bs - 1) / bs);
      spmm_csr4<<<g, bs, 0, stream>>>(outp, inp, rowptr, csr_s, csr_w, subp, coef, fin);
    } else {
      dim3 g(N_NODES, (fin + 127) / 128);
      spmm_csr<<<g, 128, 0, stream>>>(outp, inp, rowptr, csr_s, csr_w, subp, coef, fin);
    }
  };

  const int fins[4]  = {103, 192, 384, 768};
  const int fouts[4] = {64, 128, 256, 256};
  const int KL[3] = {2, 4, 6};

  const float* X = x;     // layer input (T0)
  float* Hc = hA;         // layer output
  for (int l = 0; l < 4; ++l) {
    const int fin = fins[l], fo = fouts[l], ldh = 3 * fo;
    dim3 ggrid(fo / 64, (N_NODES + 127) / 128);

    // k = 0: H_s = T0 @ W_s[0] + b_s   (initializes H with bias)
    for (int s = 0; s < 3; ++s) {
      const float* Wp = (const float*)d_in[wIdx[l] + s];
      const float* Bp = (const float*)d_in[bIdx[l] + s];
      gemm_bf16_wmma<<<ggrid, 256, 0, stream>>>(X, Wp, Bp, Hc, ldh, s * fo, fin, fo, 0);
    }

    // T1 = L~ T0
    launch_spmm(TB[0], X, nullptr, 1.0f, fin);

    const float* p0 = X;        // T_{k-1}
    const float* p1 = TB[0];    // T_k
    for (int k = 1; k <= 5; ++k) {
      for (int s = 0; s < 3; ++s) {
        if (KL[s] > k) {
          const float* Wp = (const float*)d_in[wIdx[l] + s] + (size_t)k * fin * fo;
          gemm_bf16_wmma<<<ggrid, 256, 0, stream>>>(p1, Wp, nullptr, Hc, ldh, s * fo, fin, fo, 1);
        }
      }
      if (k < 5) {
        // T_{k+1} = 2 * L~ T_k - T_{k-1}   (fused into SpMM epilogue)
        float* pn = TB[k % 3];   // rotation: k=1->TB1, 2->TB2, 3->TB0, 4->TB1
        launch_spmm(pn, p1, p0, 2.0f, fin);
        p0 = p1; p1 = pn;
      }
    }

    size_t hcnt4 = ((size_t)N_NODES * ldh) >> 2;
    relu_f32v4<<<((hcnt4 + 255) / 256), 256, 0, stream>>>((float4*)Hc, hcnt4);
    X = Hc;
    Hc = (Hc == hA) ? hB : hA;
  }

  fc_kernel<<<(N_NODES + 255) / 256, 256, 0, stream>>>(X, fc_w, fc_b, out);
}
